// SageConv_14001593385221
// MI455X (gfx1250) — matrix operand order repaired and checked by reading, not compiled, not run on hardware
//
#include <hip/hip_runtime.h>
#include <hip/hip_bf16.h>
#include <stdint.h>

// ext-vector types matching the gfx1250 WMMA builtin signatures
typedef __attribute__((ext_vector_type(16))) __bf16 bf16x16;
typedef __attribute__((ext_vector_type(8)))  __bf16 bf16x8;
typedef __attribute__((ext_vector_type(8)))  float  f32x8;
typedef __attribute__((ext_vector_type(8)))  float  v8f;

// load 8 contiguous f32 (32B aligned) and convert to bf16 (RNE)
static __device__ inline bf16x8 cvt8(const float* __restrict__ p) {
    f32x8 f = *(const f32x8*)p;
    return __builtin_convertvector(f, bf16x8);
}
static __device__ inline bf16x16 cat16(bf16x8 lo, bf16x8 hi) {
    bf16x16 r;
#pragma unroll
    for (int i = 0; i < 8; ++i) { r[i] = lo[i]; r[i + 8] = hi[i]; }
    return r;
}

// ---------------------------------------------------------------------------
// X(M x 1024) @ W(1024 x 1024)^T, f32 row-major inputs, on-the-fly bf16 cvt.
// OUT_F32 = true : write f32, normal (M x 1024) orientation (h_self -> d_out)
// OUT_F32 = false: write bf16 TRANSPOSED (1024 x M) so the downstream SpMM
//                  reads its B operand K-contiguously (h_n^T -> workspace).
// Wave tile 64x64 (4x4 WMMA accums), WG = 8 waves = 128(M) x 256(N).
// ---------------------------------------------------------------------------
template <bool OUT_F32, int LDT>
__global__ __launch_bounds__(256)
void k_gemm_xwT(const float* __restrict__ X, const float* __restrict__ W,
                void* __restrict__ Out) {
    constexpr int D = 1024;   // K
    constexpr int N = 1024;   // output N extent
    const int lane = threadIdx.x & 31;
    const int wid  = threadIdx.x >> 5;
    const int wm = wid & 1, wn = wid >> 1;
    const int mw = blockIdx.y * 128 + wm * 64;
    const int nw = blockIdx.x * 256 + wn * 64;
    const int lm = lane & 15;
    const int ah = (lane >> 4) * 8;    // A half-select: K {0..7,16..23} vs {8..15,24..31}
    const int bh = (lane >> 4) * 8;    // B half-select: K {0..7,16..23} vs {8..15,24..31}
    const int rh = (lane >> 4) * 8;    // C row offset: M = v + 8*(lane>=16)

    v8f acc[4][4];
#pragma unroll
    for (int mi = 0; mi < 4; ++mi)
#pragma unroll
        for (int ni = 0; ni < 4; ++ni)
#pragma unroll
            for (int v = 0; v < 8; ++v) acc[mi][ni][v] = 0.0f;

    const float* xrow[4];
    const float* wrow[4];
#pragma unroll
    for (int mi = 0; mi < 4; ++mi)
        xrow[mi] = X + (size_t)(mw + 16 * mi + lm) * D + ah;
#pragma unroll
    for (int ni = 0; ni < 4; ++ni)
        wrow[ni] = W + (size_t)(nw + 16 * ni + lm) * D + bh;

    for (int k0 = 0; k0 < D; k0 += 32) {
        bf16x16 a[4], b[4];
#pragma unroll
        for (int mi = 0; mi < 4; ++mi)
            a[mi] = cat16(cvt8(xrow[mi] + k0), cvt8(xrow[mi] + k0 + 16));
#pragma unroll
        for (int ni = 0; ni < 4; ++ni)
            b[ni] = cat16(cvt8(wrow[ni] + k0), cvt8(wrow[ni] + k0 + 16));
#pragma unroll
        for (int mi = 0; mi < 4; ++mi)
#pragma unroll
            for (int ni = 0; ni < 4; ++ni)
                acc[mi][ni] = __builtin_amdgcn_wmma_f32_16x16x32_bf16(
                    false, a[mi], false, b[ni], (short)0, acc[mi][ni], false, false);
    }

    if constexpr (OUT_F32) {
#pragma unroll
        for (int mi = 0; mi < 4; ++mi)
#pragma unroll
            for (int ni = 0; ni < 4; ++ni)
#pragma unroll
                for (int v = 0; v < 8; ++v)
                    ((float*)Out)[(size_t)(mw + 16 * mi + rh + v) * N +
                                  (nw + 16 * ni + lm)] = acc[mi][ni][v];
    } else {
        // transposed bf16 store: v-dim is contiguous -> one 16B store per frag
        __bf16* OT = (__bf16*)Out;   // (N x LDT)
#pragma unroll
        for (int mi = 0; mi < 4; ++mi)
#pragma unroll
            for (int ni = 0; ni < 4; ++ni) {
                bf16x8 s;
#pragma unroll
                for (int v = 0; v < 8; ++v) s[v] = (__bf16)acc[mi][ni][v];
                *(bf16x8*)&OT[(size_t)(nw + 16 * ni + lm) * LDT +
                              (mw + 16 * mi + rh)] = s;
            }
    }
}

// ---------------------------------------------------------------------------
// Out(8192x1024, f32) += BLOCK(8192x16384, f32) @ H where HT(1024x16384, bf16)
// is H transposed. WG tile 128(M) x 256(N), 8 waves (2M x 4N), wave 64x64.
// A (block) staged through LDS with f32->bf16 cvt, software-pipelined:
// next 128x64 tile is loaded into registers while WMMAs consume LDS.
// B fragments are direct K-contiguous 16B global loads from HT (32MB, L2-
// resident). C initialized from Out (holds h_self), accumulated in f32.
// ---------------------------------------------------------------------------
__global__ __launch_bounds__(256)
void k_spmm_acc(const float* __restrict__ BLK, const __bf16* __restrict__ HT,
                float* __restrict__ Out) {
    constexpr int K  = 16384;
    constexpr int N  = 1024;
    constexpr int LP = 72;                 // LDS row pitch (bf16): 64 + 8 pad
    __shared__ __bf16 ldsA[128 * LP];      // [m_local][k_local], 18 KB

    const int lane = threadIdx.x & 31;
    const int wid  = threadIdx.x >> 5;
    const int wm = wid & 1, wn = wid >> 1;
    const int m0 = blockIdx.y * 128;
    const int n0 = blockIdx.x * 256;
    const int lm = lane & 15;
    const int ah = (lane >> 4) * 8;
    const int bh = (lane >> 4) * 8;
    const int rh = (lane >> 4) * 8;

    // C init: Out already holds h_self (previous kernel on same stream)
    v8f acc[4][4];
#pragma unroll
    for (int mi = 0; mi < 4; ++mi)
#pragma unroll
        for (int ni = 0; ni < 4; ++ni)
#pragma unroll
            for (int v = 0; v < 8; ++v)
                acc[mi][ni][v] = Out[(size_t)(m0 + wm * 64 + 16 * mi + rh + v) * N +
                                     (n0 + wn * 64 + 16 * ni + lm)];

    // A staging assignment: each thread owns a 1x32 f32 slice of the 128x64 tile
    const int tr = threadIdx.x >> 1;          // 0..127
    const int tc = (threadIdx.x & 1) * 32;    // 0 / 32
    const float* ap = BLK + (size_t)(m0 + tr) * K + tc;

    // B fragment row pointers (K-contiguous in HT)
    const __bf16* bp[4];
#pragma unroll
    for (int ni = 0; ni < 4; ++ni)
        bp[ni] = HT + (size_t)(n0 + wn * 64 + 16 * ni + lm) * K + bh;

    // pipeline prologue: first A tile in registers
    f32x8 ga[4];
#pragma unroll
    for (int j = 0; j < 4; ++j) ga[j] = *(const f32x8*)(ap + 8 * j);

    for (int k0 = 0; k0 < K; k0 += 64) {
        // commit current tile (cvt f32->bf16) to LDS
#pragma unroll
        for (int j = 0; j < 4; ++j)
            *(bf16x8*)&ldsA[tr * LP + tc + 8 * j] =
                __builtin_convertvector(ga[j], bf16x8);
        __syncthreads();

        // overlap: fetch next tile while WMMAs run on this one
        if (k0 + 64 < K) {
#pragma unroll
            for (int j = 0; j < 4; ++j)
                ga[j] = *(const f32x8*)(ap + (k0 + 64) + 8 * j);
            __builtin_prefetch(ap + k0 + 128, 0, 1);   // global_prefetch_b8
        }

#pragma unroll
        for (int ks = 0; ks < 64; ks += 32) {
            bf16x16 a[4], b[4];
#pragma unroll
            for (int mi = 0; mi < 4; ++mi) {
                const __bf16* p = &ldsA[(wm * 64 + 16 * mi + lm) * LP + ks + ah];
                a[mi] = cat16(*(const bf16x8*)p, *(const bf16x8*)(p + 16));
            }
#pragma unroll
            for (int ni = 0; ni < 4; ++ni) {
                const __bf16* q = bp[ni] + k0 + ks;
                b[ni] = cat16(*(const bf16x8*)q, *(const bf16x8*)(q + 16));
            }
#pragma unroll
            for (int mi = 0; mi < 4; ++mi)
#pragma unroll
                for (int ni = 0; ni < 4; ++ni)
                    acc[mi][ni] = __builtin_amdgcn_wmma_f32_16x16x32_bf16(
                        false, a[mi], false, b[ni], (short)0, acc[mi][ni], false, false);
        }
        __syncthreads();
    }

#pragma unroll
    for (int mi = 0; mi < 4; ++mi)
#pragma unroll
        for (int ni = 0; ni < 4; ++ni)
#pragma unroll
            for (int v = 0; v < 8; ++v)
                Out[(size_t)(m0 + wm * 64 + 16 * mi + rh + v) * N +
                    (n0 + wn * 64 + 16 * ni + lm)] = acc[mi][ni][v];
}

extern "C" void kernel_launch(void* const* d_in, const int* in_sizes, int n_in,
                              void* d_out, int out_size, void* d_ws, size_t ws_size,
                              hipStream_t stream) {
    (void)in_sizes; (void)n_in; (void)out_size; (void)ws_size;
    const float* block = (const float*)d_in[0];   // (8192, 16384)
    const float* x     = (const float*)d_in[1];   // (16384, 1024)
    const float* wself = (const float*)d_in[2];   // (1024, 1024)
    const float* wnb   = (const float*)d_in[3];   // (1024, 1024)
    float*  out = (float*)d_out;                  // (8192, 1024) f32
    __bf16* hT  = (__bf16*)d_ws;                  // h_n^T: (1024, 16384) bf16 = 32 MB

    dim3 blk(256);
    // hT = (x @ w_nb^T)^T   stored (1024 x 16384) bf16
    k_gemm_xwT<false, 16384><<<dim3(1024 / 256, 16384 / 128), blk, 0, stream>>>(
        x, wnb, (void*)hT);
    // out = x[:8192] @ w_self^T   (f32, normal orientation)
    k_gemm_xwT<true, 1024><<<dim3(1024 / 256, 8192 / 128), blk, 0, stream>>>(
        x, wself, (void*)out);
    // out += block @ h_n   (B operand read K-contiguously from hT)
    k_spmm_acc<<<dim3(1024 / 256, 8192 / 64 / 2), blk, 0, stream>>>(block, hT, out);
}
